// InfiniHRM_45457933861049
// MI455X (gfx1250) — compile-verified, matrix operand order checked by reading, not run on hardware
//
#include <hip/hip_runtime.h>
#include <hip/hip_bf16.h>

// ---------------------------------------------------------------------------
// InfiniHRM forward for MI455X (gfx1250, wave32, WMMA).
// All GEMMs: bf16 inputs, f32 accumulate via v_wmma_f32_16x16x32_bf16.
// Tile staging via GLOBAL_LOAD_ASYNC_TO_LDS_B128 + double buffering when the
// toolchain exposes the builtin; falls back to synchronous LDS staging.
// ---------------------------------------------------------------------------

#define BATCH  4
#define SEQ    1024
#define DIM    1024
#define HEADS  16
#define HD     64
#define VOCAB  32000
#define MROWS  (BATCH * SEQ)          // 4096
#define NELEM  (MROWS * DIM)          // 4096*1024

typedef __attribute__((ext_vector_type(16))) __bf16 v16bf;
typedef __attribute__((ext_vector_type(8)))  float  v8f;

union FragU { v16bf bf; uint4 u[2]; };
union AccU  { v8f v; float f[8]; };

// ---- gfx1250 async global->LDS path (guarded) -----------------------------
#if defined(__has_builtin)
#if __has_builtin(__builtin_amdgcn_global_load_async_to_lds_b128)
#define HAVE_ASYNC_LDS 1
#endif
#endif

#if defined(HAVE_ASYNC_LDS)
typedef int v4i __attribute__((vector_size(16)));
typedef __attribute__((address_space(1))) v4i gas_v4i;   // global (prints as __device__)
typedef __attribute__((address_space(3))) v4i las_v4i;   // LDS   (prints as __shared__)
__device__ __forceinline__ void async_cp16(const unsigned short* gsrc, unsigned short* lds) {
    __builtin_amdgcn_global_load_async_to_lds_b128((gas_v4i*)gsrc, (las_v4i*)lds, 0, 0);
}
#if __has_builtin(__builtin_amdgcn_s_wait_asynccnt)
#define WAIT_ASYNCCNT(n) __builtin_amdgcn_s_wait_asynccnt(n)
#else
#define WAIT_ASYNCCNT(n) asm volatile("s_wait_asynccnt " #n ::: "memory")
#endif
#endif

__device__ __forceinline__ unsigned short f2bf(float f) {
    unsigned int u = __float_as_uint(f);
    u = u + 0x7FFFu + ((u >> 16) & 1u);          // round to nearest even
    return (unsigned short)(u >> 16);
}
__device__ __forceinline__ float geluf(float x) {
    const float k0 = 0.7978845608028654f;
    return 0.5f * x * (1.0f + tanhf(k0 * (x + 0.044715f * x * x * x)));
}

// Load one 16x32 bf16 fragment (A or B operand) from an LDS tile row.
// Per-lane layout (CDNA5 §7.12.2): lane L (m = L&15, kh = L>>4) holds
// K = kbase + {kh*8 .. kh*8+7} and K = kbase + {16+kh*8 .. 16+kh*8+7}.
__device__ __forceinline__ v16bf ld_frag(const unsigned short* row_base, int k4, int kh) {
    FragU f;
    const uint4* p = (const uint4*)row_base;
    f.u[0] = p[k4 + kh];
    f.u[1] = p[k4 + 2 + kh];
    return f.bf;
}

// ---------------------------------------------------------------------------
// Generic GEMM: C[M,N] = A[M,K](bf16) @ W[N,K](bf16)^T   (nn.Linear layout)
// mode bit0: write f32 C, bit1: write bf16 C, bit2: GELU epilogue.
// Tile 128x128, 256 threads = 8 waves, each wave 32x64 (2x4 wmma tiles).
// ---------------------------------------------------------------------------
__global__ __launch_bounds__(256) void k_gemm_bf16(
    const unsigned short* __restrict__ A, const unsigned short* __restrict__ W,
    float* __restrict__ Cf, unsigned short* __restrict__ Cb,
    int M, int N, int K, int mode)
{
    const int tid  = threadIdx.x;
    const int lane = tid & 31;
    const int wid  = tid >> 5;
    const int wm   = wid & 3;          // 4 waves along M
    const int wn   = wid >> 2;         // 2 waves along N
    const int lm   = lane & 15;
    const int kh   = lane >> 4;
    const int m0   = blockIdx.y * 128;
    const int n0   = blockIdx.x * 128;

    AccU acc[2][4] = {};

#if defined(HAVE_ASYNC_LDS)
    // Double-buffered async staging: the async engine fills tile (it+1)
    // while the waves run WMMAs on tile (it).
    __shared__ __align__(16) unsigned short As[2][128 * 32];
    __shared__ __align__(16) unsigned short Bs[2][128 * 32];

    const int r0  = (tid + 0)   >> 2, c0 = ((tid + 0)   & 3) * 8;
    const int r1  = (tid + 256) >> 2, c1 = ((tid + 256) & 3) * 8;

    // issue tile 0 (4 async-load instructions per thread)
    async_cp16(&A[(size_t)(m0 + r0) * K + c0], &As[0][r0 * 32 + c0]);
    async_cp16(&W[(size_t)(n0 + r0) * K + c0], &Bs[0][r0 * 32 + c0]);
    async_cp16(&A[(size_t)(m0 + r1) * K + c1], &As[0][r1 * 32 + c1]);
    async_cp16(&W[(size_t)(n0 + r1) * K + c1], &Bs[0][r1 * 32 + c1]);

    const int nk = K >> 5;
    for (int it = 0; it < nk; ++it) {
        const int cur = it & 1;
        if (it + 1 < nk) {
            const size_t kn = (size_t)(it + 1) * 32;
            async_cp16(&A[(size_t)(m0 + r0) * K + kn + c0], &As[cur ^ 1][r0 * 32 + c0]);
            async_cp16(&W[(size_t)(n0 + r0) * K + kn + c0], &Bs[cur ^ 1][r0 * 32 + c0]);
            async_cp16(&A[(size_t)(m0 + r1) * K + kn + c1], &As[cur ^ 1][r1 * 32 + c1]);
            async_cp16(&W[(size_t)(n0 + r1) * K + kn + c1], &Bs[cur ^ 1][r1 * 32 + c1]);
            WAIT_ASYNCCNT(4);           // oldest 4 (tile `cur`) complete, next 4 in flight
        } else {
            WAIT_ASYNCCNT(0);
        }
        __syncthreads();                // all waves' slices of tile `cur` visible

        v16bf af[2], bfv[4];
        #pragma unroll
        for (int mt = 0; mt < 2; ++mt)
            af[mt] = ld_frag(&As[cur][(wm * 32 + mt * 16 + lm) * 32], 0, kh);
        #pragma unroll
        for (int nt = 0; nt < 4; ++nt)
            bfv[nt] = ld_frag(&Bs[cur][(wn * 64 + nt * 16 + lm) * 32], 0, kh);

        #pragma unroll
        for (int mt = 0; mt < 2; ++mt)
            #pragma unroll
            for (int nt = 0; nt < 4; ++nt)
                acc[mt][nt].v = __builtin_amdgcn_wmma_f32_16x16x32_bf16(
                    false, af[mt], false, bfv[nt], (short)0, acc[mt][nt].v,
                    false, false);

        __syncthreads();                // done reading `cur` before it is refilled
    }
#else
    // Fallback: synchronous single-buffer staging.
    __shared__ __align__(16) unsigned short As[128 * 32];
    __shared__ __align__(16) unsigned short Bs[128 * 32];

    for (int k0 = 0; k0 < K; k0 += 32) {
        __syncthreads();
        #pragma unroll
        for (int i = 0; i < 2; ++i) {
            int idx = tid + i * 256;
            int row = idx >> 2;
            int cb  = (idx & 3) * 8;
            *(uint4*)&As[row * 32 + cb] =
                *(const uint4*)&A[(size_t)(m0 + row) * K + k0 + cb];
            *(uint4*)&Bs[row * 32 + cb] =
                *(const uint4*)&W[(size_t)(n0 + row) * K + k0 + cb];
        }
        __syncthreads();

        v16bf af[2], bfv[4];
        #pragma unroll
        for (int mt = 0; mt < 2; ++mt)
            af[mt] = ld_frag(&As[(wm * 32 + mt * 16 + lm) * 32], 0, kh);
        #pragma unroll
        for (int nt = 0; nt < 4; ++nt)
            bfv[nt] = ld_frag(&Bs[(wn * 64 + nt * 16 + lm) * 32], 0, kh);

        #pragma unroll
        for (int mt = 0; mt < 2; ++mt)
            #pragma unroll
            for (int nt = 0; nt < 4; ++nt)
                acc[mt][nt].v = __builtin_amdgcn_wmma_f32_16x16x32_bf16(
                    false, af[mt], false, bfv[nt], (short)0, acc[mt][nt].v,
                    false, false);
    }
#endif

    #pragma unroll
    for (int mt = 0; mt < 2; ++mt)
        #pragma unroll
        for (int nt = 0; nt < 4; ++nt)
            #pragma unroll
            for (int r = 0; r < 8; ++r) {
                int m = m0 + wm * 32 + mt * 16 + kh * 8 + r;
                int n = n0 + wn * 64 + nt * 16 + lm;
                float v = acc[mt][nt].f[r];
                if (mode & 4) v = geluf(v);
                if (mode & 1) Cf[(size_t)m * N + n] = v;
                if (mode & 2) Cb[(size_t)m * N + n] = f2bf(v);
            }
}

// ---------------------------------------------------------------------------
// Flash attention, one block = one (b,h) x one 64-row q-tile. 4 waves,
// each wave owns 16 q-rows. WMMA for QK^T and PV; online softmax in f32.
// gate != nullptr: scale output by (1 - sigmoid(gate[h])) (Infini, memory=0).
// ---------------------------------------------------------------------------
__global__ __launch_bounds__(128) void k_attn(
    const unsigned short* __restrict__ Qg, const unsigned short* __restrict__ Kg,
    const unsigned short* __restrict__ Vg, unsigned short* __restrict__ Og,
    const float* __restrict__ gate, int causal)
{
    const int qt  = blockIdx.x;              // 0..15
    const int bh  = blockIdx.y;              // 0..63
    const int b   = bh >> 4, h = bh & 15;
    const int tid = threadIdx.x;
    const int lane = tid & 31, w = tid >> 5;
    const int lm = lane & 15, kh = lane >> 4;

    __shared__ __align__(16) unsigned short Qs[64 * 64];
    __shared__ __align__(16) unsigned short Ks[64 * 64];
    __shared__ __align__(16) unsigned short Vt[64 * 64];   // transposed: [d][kpos]
    __shared__ __align__(16) unsigned short Ps[64 * 64];

    const size_t base = ((size_t)b * SEQ) * DIM + (size_t)h * HD;

    #pragma unroll
    for (int i = 0; i < 4; ++i) {
        int idx = tid + i * 128;
        int row = idx >> 3, c = (idx & 7) * 8;
#if defined(HAVE_ASYNC_LDS)
        async_cp16(&Qg[base + (size_t)(qt * 64 + row) * DIM + c], &Qs[row * 64 + c]);
#else
        *(uint4*)&Qs[row * 64 + c] =
            *(const uint4*)&Qg[base + (size_t)(qt * 64 + row) * DIM + c];
#endif
    }

    float mx[8], sm[8];
    AccU o4[4] = {};
    #pragma unroll
    for (int r = 0; r < 8; ++r) { mx[r] = -__builtin_inff(); sm[r] = 0.f; }

    const int ktmax = causal ? qt : (SEQ / 64 - 1);
    const int qrow_base = qt * 64 + w * 16 + kh * 8;

    for (int kt = 0; kt <= ktmax; ++kt) {
        __syncthreads();
        #pragma unroll
        for (int i = 0; i < 4; ++i) {
            int idx = tid + i * 128;
            int row = idx >> 3, c = (idx & 7) * 8;
#if defined(HAVE_ASYNC_LDS)
            async_cp16(&Kg[base + (size_t)(kt * 64 + row) * DIM + c], &Ks[row * 64 + c]);
#else
            *(uint4*)&Ks[row * 64 + c] =
                *(const uint4*)&Kg[base + (size_t)(kt * 64 + row) * DIM + c];
#endif
            union { uint4 u; unsigned short s[8]; } vv;
            vv.u = *(const uint4*)&Vg[base + (size_t)(kt * 64 + row) * DIM + c];
            #pragma unroll
            for (int j = 0; j < 8; ++j) Vt[(c + j) * 64 + row] = vv.s[j];
        }
#if defined(HAVE_ASYNC_LDS)
        WAIT_ASYNCCNT(0);
#endif
        __syncthreads();

        // S = Q K^T  (K-dim 64 -> two wmma steps)
        v16bf a0 = ld_frag(&Qs[(w * 16 + lm) * 64], 0, kh);
        v16bf a1 = ld_frag(&Qs[(w * 16 + lm) * 64], 4, kh);
        AccU s4[4] = {};
        #pragma unroll
        for (int nt = 0; nt < 4; ++nt) {
            v16bf b0 = ld_frag(&Ks[(nt * 16 + lm) * 64], 0, kh);
            v16bf b1 = ld_frag(&Ks[(nt * 16 + lm) * 64], 4, kh);
            s4[nt].v = __builtin_amdgcn_wmma_f32_16x16x32_bf16(
                false, a0, false, b0, (short)0, s4[nt].v, false, false);
            s4[nt].v = __builtin_amdgcn_wmma_f32_16x16x32_bf16(
                false, a1, false, b1, (short)0, s4[nt].v, false, false);
        }

        // online softmax per q-row (row r lives in lanes of this 16-half)
        const float sc = 0.125f;                       // HD^-0.5
        #pragma unroll
        for (int r = 0; r < 8; ++r) {
            float vmax = -__builtin_inff();
            #pragma unroll
            for (int nt = 0; nt < 4; ++nt) {
                float sv = s4[nt].f[r] * sc;
                if (causal && kt == qt) {
                    int col = kt * 64 + nt * 16 + lm;
                    if (col > qrow_base + r) sv = -__builtin_inff();
                }
                s4[nt].f[r] = sv;
                vmax = fmaxf(vmax, sv);
            }
            #pragma unroll
            for (int off = 1; off < 16; off <<= 1)
                vmax = fmaxf(vmax, __shfl_xor(vmax, off));
            float mnew  = fmaxf(mx[r], vmax);
            float scale = expf(mx[r] - mnew);
            float rs = 0.f;
            #pragma unroll
            for (int nt = 0; nt < 4; ++nt) {
                float pv = expf(s4[nt].f[r] - mnew);
                s4[nt].f[r] = pv;
                rs += pv;
            }
            #pragma unroll
            for (int off = 1; off < 16; off <<= 1)
                rs += __shfl_xor(rs, off);
            sm[r] = sm[r] * scale + rs;
            mx[r] = mnew;
            #pragma unroll
            for (int dt = 0; dt < 4; ++dt) o4[dt].f[r] *= scale;
        }

        // stage P (bf16) in this wave's private 16 LDS rows, then O += P V
        #pragma unroll
        for (int nt = 0; nt < 4; ++nt)
            #pragma unroll
            for (int r = 0; r < 8; ++r)
                Ps[(w * 16 + kh * 8 + r) * 64 + nt * 16 + lm] = f2bf(s4[nt].f[r]);

        v16bf pa0 = ld_frag(&Ps[(w * 16 + lm) * 64], 0, kh);
        v16bf pa1 = ld_frag(&Ps[(w * 16 + lm) * 64], 4, kh);
        #pragma unroll
        for (int dt = 0; dt < 4; ++dt) {
            v16bf vb0 = ld_frag(&Vt[(dt * 16 + lm) * 64], 0, kh);
            v16bf vb1 = ld_frag(&Vt[(dt * 16 + lm) * 64], 4, kh);
            o4[dt].v = __builtin_amdgcn_wmma_f32_16x16x32_bf16(
                false, pa0, false, vb0, (short)0, o4[dt].v, false, false);
            o4[dt].v = __builtin_amdgcn_wmma_f32_16x16x32_bf16(
                false, pa1, false, vb1, (short)0, o4[dt].v, false, false);
        }
    }

    float g = 1.f;
    if (gate) g = 1.f / (1.f + expf(gate[h]));        // 1 - sigmoid(gate)
    #pragma unroll
    for (int dt = 0; dt < 4; ++dt)
        #pragma unroll
        for (int r = 0; r < 8; ++r) {
            float val = o4[dt].f[r] / sm[r] * g;
            Og[base + (size_t)(qrow_base + r) * DIM + dt * 16 + lm] = f2bf(val);
        }
}

// ---------------------------------------------------------------------------
// Elementwise / reduction helpers
// ---------------------------------------------------------------------------
__global__ void k_cast(const float* __restrict__ a, unsigned short* __restrict__ o, int n) {
    int i = blockIdx.x * blockDim.x + threadIdx.x;
    if (i < n) o[i] = f2bf(a[i]);
}
__global__ void k_embed(const int* __restrict__ x, const float* __restrict__ emb,
                        unsigned short* __restrict__ o, int n) {
    int i = blockIdx.x * blockDim.x + threadIdx.x;
    if (i < n) {
        int tok = x[i / DIM];
        o[i] = f2bf(emb[(size_t)tok * DIM + (i % DIM)]);
    }
}
__global__ void k_bcast(const float* __restrict__ init, float* __restrict__ of,
                        unsigned short* __restrict__ ob, int n, int srcn) {
    int i = blockIdx.x * blockDim.x + threadIdx.x;
    if (i < n) { float v = init[i % srcn]; of[i] = v; ob[i] = f2bf(v); }
}
__global__ void k_add3(const float* __restrict__ a, const float* __restrict__ b,
                       const float* __restrict__ c, float* __restrict__ of,
                       unsigned short* __restrict__ ob, int n) {
    int i = blockIdx.x * blockDim.x + threadIdx.x;
    if (i < n) { float v = a[i] + b[i] + c[i]; of[i] = v; ob[i] = f2bf(v); }
}
// out = rmsnorm(a + r), one block per row of DIM
__global__ __launch_bounds__(256) void k_addnorm(
    const float* __restrict__ a, const float* __restrict__ r,
    float* __restrict__ of, unsigned short* __restrict__ ob)
{
    __shared__ float red[8];
    const size_t o = (size_t)blockIdx.x * DIM;
    const int tid = threadIdx.x;
    float t[4]; float ss = 0.f;
    #pragma unroll
    for (int i = 0; i < 4; ++i) {
        int c = tid + i * 256;
        t[i] = a[o + c] + r[o + c];
        ss += t[i] * t[i];
    }
    #pragma unroll
    for (int off = 1; off < 32; off <<= 1) ss += __shfl_xor(ss, off);
    if ((tid & 31) == 0) red[tid >> 5] = ss;
    __syncthreads();
    float tot = 0.f;
    #pragma unroll
    for (int i = 0; i < 8; ++i) tot += red[i];
    float rms = rsqrtf(tot / DIM + 1e-6f);
    #pragma unroll
    for (int i = 0; i < 4; ++i) {
        int c = tid + i * 256;
        float v = t[i] * rms;
        of[o + c] = v; ob[o + c] = f2bf(v);
    }
}
__global__ void k_meanseq(const float* __restrict__ s, float* __restrict__ mean) {
    int b = blockIdx.x;
    for (int d = threadIdx.x; d < DIM; d += blockDim.x) {
        float acc = 0.f;
        for (int t = 0; t < SEQ; ++t) acc += s[((size_t)b * SEQ + t) * DIM + d];
        mean[b * DIM + d] = acc * (1.0f / SEQ);
    }
}
__global__ void k_qhead(const float* __restrict__ mean, const float* __restrict__ wq,
                        float* __restrict__ out) {
    int i = threadIdx.x;
    if (i < BATCH * 2) {
        int b = i >> 1, oo = i & 1;
        float acc = 0.f;
        for (int d = 0; d < DIM; ++d) acc += mean[b * DIM + d] * wq[oo * DIM + d];
        out[i] = 1.f / (1.f + expf(-acc));
    }
}

// ---------------------------------------------------------------------------
// Host orchestration
// ---------------------------------------------------------------------------
static void gemm(hipStream_t st, const unsigned short* A, const unsigned short* W,
                 float* Cf, unsigned short* Cb, int M, int N, int K, int mode) {
    dim3 g(N / 128, M / 128);
    k_gemm_bf16<<<g, 256, 0, st>>>(A, W, Cf, Cb, M, N, K, mode);
}

extern "C" void kernel_launch(void* const* d_in, const int* in_sizes, int n_in,
                              void* d_out, int out_size, void* d_ws, size_t ws_size,
                              hipStream_t stream) {
    (void)in_sizes; (void)n_in; (void)out_size; (void)ws_size;

    // ---- workspace carve-out ----
    char* ws = (char*)d_ws;
    size_t off = 0;
    auto alloc = [&](size_t bytes) -> char* {
        char* p = ws + off;
        off = (off + bytes + 255) & ~(size_t)255;
        return p;
    };
    const size_t SW = (size_t)DIM * DIM;            // 1M elems small weight
    const size_t FW = (size_t)4 * DIM * DIM;        // 4M elems ffn weight

    unsigned short* wsm[12];                         // a_{q,k,v,o}, l0_{q,k,v,o}, l1_{q,k,v,o}
    for (int i = 0; i < 12; ++i) wsm[i] = (unsigned short*)alloc(SW * 2);
    unsigned short* wff[4];                          // l0_w1,l0_w2,l1_w1,l1_w2
    for (int i = 0; i < 4; ++i)  wff[i] = (unsigned short*)alloc(FW * 2);
    unsigned short* woutb = (unsigned short*)alloc((size_t)VOCAB * DIM * 2);

    unsigned short* Xb  = (unsigned short*)alloc((size_t)NELEM * 2);
    float* xenr         = (float*)alloc((size_t)NELEM * 4);
    float* s0f          = (float*)alloc((size_t)NELEM * 4);
    unsigned short* s0b = (unsigned short*)alloc((size_t)NELEM * 2);
    float* s1f          = (float*)alloc((size_t)NELEM * 4);
    unsigned short* s1b = (unsigned short*)alloc((size_t)NELEM * 2);
    float* hf           = (float*)alloc((size_t)NELEM * 4);
    unsigned short* hb  = (unsigned short*)alloc((size_t)NELEM * 2);
    float* h1f          = (float*)alloc((size_t)NELEM * 4);
    unsigned short* h1b = (unsigned short*)alloc((size_t)NELEM * 2);
    float* h2f          = (float*)alloc((size_t)NELEM * 4);
    unsigned short* h2b = (unsigned short*)alloc((size_t)NELEM * 2);
    unsigned short* Qb  = (unsigned short*)alloc((size_t)NELEM * 2);
    unsigned short* Kb  = (unsigned short*)alloc((size_t)NELEM * 2);
    unsigned short* Vb  = (unsigned short*)alloc((size_t)NELEM * 2);
    unsigned short* Ab  = (unsigned short*)alloc((size_t)NELEM * 2);
    float* t1           = (float*)alloc((size_t)NELEM * 4);
    unsigned short* Gb  = (unsigned short*)alloc((size_t)MROWS * 4 * DIM * 2);
    float* meanb        = (float*)alloc((size_t)BATCH * DIM * 4);

    // ---- inputs (setup_inputs order) ----
    const int*   x     = (const int*)d_in[0];
    const float* emb   = (const float*)d_in[1];
    const float* gate  = (const float*)d_in[6];
    const float* init0 = (const float*)d_in[7];
    const float* init1 = (const float*)d_in[8];
    const float* wqh   = (const float*)d_in[22];

    // ---- weight casts to bf16 ----
    const int widx[12] = {2, 3, 4, 5, 9, 10, 11, 12, 15, 16, 17, 18};
    for (int i = 0; i < 12; ++i)
        k_cast<<<(int)((SW + 255) / 256), 256, 0, stream>>>((const float*)d_in[widx[i]], wsm[i], (int)SW);
    const int fidx[4] = {13, 14, 19, 20};
    for (int i = 0; i < 4; ++i)
        k_cast<<<(int)((FW + 255) / 256), 256, 0, stream>>>((const float*)d_in[fidx[i]], wff[i], (int)FW);
    {
        const size_t n = (size_t)VOCAB * DIM;
        k_cast<<<(int)((n + 255) / 256), 256, 0, stream>>>((const float*)d_in[21], woutb, (int)n);
    }

    const int EB = (NELEM + 255) / 256;

    // ---- embedding + state init ----
    k_embed<<<EB, 256, 0, stream>>>(x, emb, Xb, NELEM);
    k_bcast<<<EB, 256, 0, stream>>>(init0, s0f, s0b, NELEM, SEQ * DIM);
    k_bcast<<<EB, 256, 0, stream>>>(init1, s1f, s1b, NELEM, SEQ * DIM);

    // ---- Infini-attention (memory==0 -> gated causal attention) ----
    gemm(stream, Xb, wsm[0], nullptr, Qb, MROWS, DIM, DIM, 2);
    gemm(stream, Xb, wsm[1], nullptr, Kb, MROWS, DIM, DIM, 2);
    gemm(stream, Xb, wsm[2], nullptr, Vb, MROWS, DIM, DIM, 2);
    k_attn<<<dim3(SEQ / 64, BATCH * HEADS), 128, 0, stream>>>(Qb, Kb, Vb, Ab, gate, 1);
    gemm(stream, Ab, wsm[3], xenr, nullptr, MROWS, DIM, DIM, 1);

    // ---- HRM layer chain ----
    auto run_layer = [&](const unsigned short* hin_b, const float* hin_f, int li,
                         float* sf_out, unsigned short* sb_out) {
        const unsigned short* wq = wsm[4 + li * 4 + 0];
        const unsigned short* wk = wsm[4 + li * 4 + 1];
        const unsigned short* wv = wsm[4 + li * 4 + 2];
        const unsigned short* wo = wsm[4 + li * 4 + 3];
        const unsigned short* w1 = wff[li * 2 + 0];
        const unsigned short* w2 = wff[li * 2 + 1];
        gemm(stream, hin_b, wq, nullptr, Qb, MROWS, DIM, DIM, 2);
        gemm(stream, hin_b, wk, nullptr, Kb, MROWS, DIM, DIM, 2);
        gemm(stream, hin_b, wv, nullptr, Vb, MROWS, DIM, DIM, 2);
        k_attn<<<dim3(SEQ / 64, BATCH * HEADS), 128, 0, stream>>>(Qb, Kb, Vb, Ab, nullptr, 0);
        gemm(stream, Ab, wo, t1, nullptr, MROWS, DIM, DIM, 1);
        k_addnorm<<<MROWS, 256, 0, stream>>>(hin_f, t1, h2f, h2b);
        gemm(stream, h2b, w1, nullptr, Gb, MROWS, 4 * DIM, DIM, 2 | 4);   // gelu, bf16
        gemm(stream, Gb, w2, t1, nullptr, MROWS, DIM, 4 * DIM, 1);
        k_addnorm<<<MROWS, 256, 0, stream>>>(h2f, t1, sf_out, sb_out);
    };

    for (int s = 0; s < 8; ++s) {
        bool dual = ((s + 1) % 4) == 0;                  // steps 3 and 7
        if (dual)   // L1 injection must read the PRE-step state0
            k_add3<<<EB, 256, 0, stream>>>(s1f, xenr, s0f, h1f, h1b, NELEM);
        k_add3<<<EB, 256, 0, stream>>>(s0f, xenr, s1f, hf, hb, NELEM);
        run_layer(hb, hf, 0, s0f, s0b);
        if (dual)
            run_layer(h1b, h1f, 1, s1f, s1b);
    }

    // ---- heads ----
    gemm(stream, s1b, woutb, (float*)d_out, nullptr, MROWS, VOCAB, DIM, 1);
    k_meanseq<<<BATCH, 256, 0, stream>>>(s1f, meanb);
    k_qhead<<<1, 64, 0, stream>>>(meanb, wqh, (float*)d_out + (size_t)MROWS * VOCAB);
}
